// GNSMsg_85186381349307
// MI455X (gfx1250) — matrix operand order repaired and checked by reading, not compiled, not run on hardware
//
#include <hip/hip_runtime.h>
#include <cmath>

// ---------------------------------------------------------------------------
// Problem constants (from reference)
// ---------------------------------------------------------------------------
#define N_NODES 4096
#define N_EDGES 131072
#define K_STEPS 30
#define D_MSG   32
// edge MLP: din=dh=35 -> pad 64 (2 x K32 chunks), dout=32
// node MLP: din=dh=68 -> pad 96 (3 x K32 chunks), dout=1 or 32

typedef __bf16 bf16;
typedef __attribute__((ext_vector_type(16))) __bf16 v16bf;
typedef __attribute__((ext_vector_type(8)))  __bf16 v8bf;
typedef __attribute__((ext_vector_type(2)))  __bf16 v2bf;
typedef __attribute__((ext_vector_type(8)))  float  v8f;

__device__ __forceinline__ v8f zero8() {
  v8f z = {0.f, 0.f, 0.f, 0.f, 0.f, 0.f, 0.f, 0.f};
  return z;
}

// D = A(16x32 bf16) * B(32x16 bf16) + C(16x16 f32)  -- CDNA5 wave32 WMMA
__device__ __forceinline__ v8f wmma_bf16(v16bf a, v16bf b, v8f c) {
  return __builtin_amdgcn_wmma_f32_16x16x32_bf16(
      /*neg_a=*/false, a, /*neg_b=*/false, b,
      /*c_mod=*/(short)0, c, /*reuse_a=*/false, /*reuse_b=*/false);
}

// A fragment (16x32 bf16) from a row-major tile (stride ld, K offset kbase).
// Contiguous-K layout used here (B packed to match): lane L holds row L%16; lanes 0-15 carry K {0..15},
// lanes 16-31 carry K {16..31}.  Each half is 8 *contiguous* bf16
// => two 16-byte loads (ds_load_b128 / global_load_b128), no scalar assembly.
__device__ __forceinline__ v16bf frag_A(const bf16* base, int ld, int kbase) {
  const int lane  = threadIdx.x & 31;
  const int row   = lane & 15;
  const int khalf = (lane & 16);       // 0 or 16
  const bf16* p = base + row * ld + kbase + khalf;
  const v8bf lo = *(const v8bf*)(p);        // K khalf..khalf+7
  const v8bf hi = *(const v8bf*)(p + 8);    // K khalf+8..khalf+15
  return __builtin_shufflevector(lo, hi, 0, 1, 2, 3, 4, 5, 6, 7,
                                         8, 9, 10, 11, 12, 13, 14, 15);
}

// B fragment from *fragment-swizzled* LDS weights laid out as
// [fragIdx][lane][16 contiguous bf16]: one aligned 32-byte vector load
// => exactly two ds_load_b128 into the WMMA source tuple.
__device__ __forceinline__ v16bf frag_Bs(const bf16* wf, int idx) {
  const int lane = threadIdx.x & 31;
  return *(const v16bf*)(wf + ((size_t)idx * 32 + lane) * 16);
}

__device__ __forceinline__ void lds_fence() {
  asm volatile("s_wait_dscnt 0" ::: "memory");
  __builtin_amdgcn_wave_barrier();
}

// ---------------------------------------------------------------------------
// Init / degree kernels
// ---------------------------------------------------------------------------
__global__ void k_init(const float* __restrict__ V0,
                       float* __restrict__ vst, float* __restrict__ thst) {
  int n = blockIdx.x * blockDim.x + threadIdx.x;
  if (n < N_NODES) { vst[n] = V0[2 * n]; thst[n] = V0[2 * n + 1]; }
}

__global__ void k_deg(const int* __restrict__ eidx, float* __restrict__ deg) {
  int e = blockIdx.x * blockDim.x + threadIdx.x;
  if (e < N_EDGES) {
    atomicAdd(&deg[eidx[2 * e]],     1.0f);
    atomicAdd(&deg[eidx[2 * e + 1]], 1.0f);
  }
}

__global__ void k_ainv(float* __restrict__ ainv) {
  int n = blockIdx.x * blockDim.x + threadIdx.x;
  if (n < N_NODES) ainv[n] = 1.0f / fmaxf(ainv[n], 1.0f);
}

__global__ void k_polar(const float* __restrict__ vst, const float* __restrict__ thst,
                        float* __restrict__ Vr, float* __restrict__ Vi) {
  int n = blockIdx.x * blockDim.x + threadIdx.x;
  if (n < N_NODES) {
    float v = vst[n], t = thst[n];
    Vr[n] = v * cosf(t);
    Vi[n] = v * sinf(t);
  }
}

// ---------------------------------------------------------------------------
// Matvec + power mismatch + loss.  One block per row; reads each Yr/Yi
// element exactly once (4 dot products per element pair).  128 MB / step,
// L2-resident after step 0 (Yr+Yi = 128 MB < 192 MB L2).
// ---------------------------------------------------------------------------
__global__ __launch_bounds__(256) void k_matvec(
    const float* __restrict__ Yr, const float* __restrict__ Yi,
    const float* __restrict__ Vr, const float* __restrict__ Vi,
    const float* __restrict__ vst, const float* __restrict__ thst,
    const float* __restrict__ Pset, const float* __restrict__ Qset,
    const int* __restrict__ btype, float* __restrict__ busf,
    float* __restrict__ loss, float wN) {
  const int row = blockIdx.x, tid = threadIdx.x;
  const float4* yr = (const float4*)(Yr + (size_t)row * N_NODES);
  const float4* yi = (const float4*)(Yi + (size_t)row * N_NODES);
  const float4* xr = (const float4*)Vr;
  const float4* xi = (const float4*)Vi;
  float srr = 0.f, sii = 0.f, sri = 0.f, sir = 0.f;
  for (int c = tid; c < N_NODES / 4; c += 256) {
    __builtin_prefetch(&yr[c + 256], 0, 0);   // global_prefetch
    __builtin_prefetch(&yi[c + 256], 0, 0);
    float4 a = yr[c], b = yi[c], x = xr[c], y = xi[c];
    srr += a.x * x.x + a.y * x.y + a.z * x.z + a.w * x.w;
    sii += b.x * y.x + b.y * y.y + b.z * y.z + b.w * y.w;
    sri += a.x * y.x + a.y * y.y + a.z * y.z + a.w * y.w;
    sir += b.x * x.x + b.y * x.y + b.z * x.z + b.w * x.w;
  }
  __shared__ float4 red[256];
  red[tid] = make_float4(srr, sii, sri, sir);
  __syncthreads();
  for (int s = 128; s > 0; s >>= 1) {
    if (tid < s) {
      float4 o = red[tid + s];
      red[tid].x += o.x; red[tid].y += o.y; red[tid].z += o.z; red[tid].w += o.w;
    }
    __syncthreads();
  }
  if (tid == 0) {
    const float Ir = red[0].x - red[0].y;        // Yr*Vr - Yi*Vi
    const float Ii = red[0].z + red[0].w;        // Yr*Vi + Yi*Vr
    const float vr = Vr[row], vi = Vi[row];
    const float P = vr * Ir + vi * Ii;
    const float Q = vi * Ir - vr * Ii;
    const int bt = btype[row];
    const bool slack = (bt == 1);
    const bool sp    = (bt == 1) || (bt == 2);
    const float dP = slack ? 0.f : Pset[row] - P;
    const float dQ = sp    ? 0.f : Qset[row] - Q;
    busf[row * 4 + 0] = vst[row];
    busf[row * 4 + 1] = thst[row];
    busf[row * 4 + 2] = dP;
    busf[row * 4 + 3] = dQ;
    atomicAdd(loss, wN * (dP * dP + dQ * dQ));
  }
}

// ---------------------------------------------------------------------------
// Edge MLP (din=dh=35 pad 64, dout=32) with bf16 WMMA + f32 LN epilogue.
// 128 threads = 4 waves; each wave owns a 16-edge tile with private LDS
// X (16x64 bf16) and H (16x64 f32) slices.  Weights staged once per block
// in fragment-swizzled order.  Layer3 scatter-adds into agg (segment_sum).
// ---------------------------------------------------------------------------
__device__ __forceinline__ void edge_layer(bf16* X, float* H, const bf16* Wf,
                                           const float* bb, const float* gg,
                                           const float* bee, int lane) {
  const v16bf a0 = frag_A(X, 64, 0);
  const v16bf a1 = frag_A(X, 64, 32);
#pragma unroll
  for (int nt = 0; nt < 4; ++nt) {
    v8f acc = zero8();
    acc = wmma_bf16(a0, frag_Bs(Wf, 0 * 4 + nt), acc);
    acc = wmma_bf16(a1, frag_Bs(Wf, 1 * 4 + nt), acc);
    const int col = nt * 16 + (lane & 15);
    const int rb  = (lane >> 4) << 3;
#pragma unroll
    for (int r = 0; r < 8; ++r) H[(rb + r) * 64 + col] = acc[r] + bb[col];
  }
  lds_fence();
  // Single-pass LayerNorm over 35 valid channels + LeakyReLU(0.1) -> bf16 X
  const int row = lane & 15;
  float s = 0.f, s2 = 0.f;
  for (int c = 0; c < 35; ++c) { const float x = H[row * 64 + c]; s += x; s2 += x * x; }
  const float mu  = s * (1.f / 35.f);
  const float var = s2 * (1.f / 35.f) - mu * mu;
  const float rs  = rsqrtf(var + 1e-5f);
  const int cb = (lane >> 4) << 5;             // lanes 0-15: cols 0..31, 16-31: 32..63
  for (int c = cb; c < cb + 32; c += 2) {
    float h0 = (c < 35)     ? ((H[row * 64 + c]     - mu) * rs * gg[c]     + bee[c])     : 0.f;
    float h1 = (c + 1 < 35) ? ((H[row * 64 + c + 1] - mu) * rs * gg[c + 1] + bee[c + 1]) : 0.f;
    h0 = (h0 > 0.f) ? h0 : 0.1f * h0;
    h1 = (h1 > 0.f) ? h1 : 0.1f * h1;
    const v2bf pr = {(bf16)h0, (bf16)h1};
    *(v2bf*)(X + row * 64 + c) = pr;
  }
  lds_fence();
}

__global__ __launch_bounds__(128) void k_edge_mlp(
    const float* __restrict__ mbuf, const int* __restrict__ eidx,
    const float* __restrict__ efeat,
    const float* W1, const float* b1, const float* g1, const float* be1,
    const float* W2, const float* b2, const float* g2, const float* be2,
    const float* W3, const float* b3, float* __restrict__ agg) {
  // Fragment-swizzled weights: [kc][nt][lane][16 elems]
  __shared__ __align__(32) bf16 sW1[2 * 4 * 32 * 16];   // 8 KB
  __shared__ __align__(32) bf16 sW2[2 * 4 * 32 * 16];   // 8 KB
  __shared__ __align__(32) bf16 sW3[2 * 2 * 32 * 16];   // 4 KB
  __shared__ float sb1[64], sg1[64], sbe1[64], sb2[64], sg2[64], sbe2[64], sb3[32];
  __shared__ __align__(32) bf16  sX[4][16 * 64];
  __shared__ __align__(32) float sH[4][16 * 64];

  const int tid = threadIdx.x;
  for (int i = tid; i < 2 * 4 * 512; i += 128) {
    const int e    = i & 15;
    const int ln   = (i >> 4) & 31;
    const int frag = i >> 9;                 // kc*4 + nt
    const int col  = (frag & 3) * 16 + (ln & 15);
    const int k    = (frag >> 2) * 32 + ((ln & 16) ? 16 : 0) + e;
    const bool v   = (k < 35) && (col < 35);
    sW1[i] = v ? (bf16)W1[k * 35 + col] : (bf16)0.f;
    sW2[i] = v ? (bf16)W2[k * 35 + col] : (bf16)0.f;
  }
  for (int i = tid; i < 2 * 2 * 512; i += 128) {
    const int e    = i & 15;
    const int ln   = (i >> 4) & 31;
    const int frag = i >> 9;                 // kc*2 + nt
    const int col  = (frag & 1) * 16 + (ln & 15);
    const int k    = (frag >> 1) * 32 + ((ln & 16) ? 16 : 0) + e;
    sW3[i] = (k < 35) ? (bf16)W3[k * 32 + col] : (bf16)0.f;
  }
  for (int i = tid; i < 64; i += 128) {
    sb1[i]  = (i < 35) ? b1[i]  : 0.f;
    sg1[i]  = (i < 35) ? g1[i]  : 0.f;
    sbe1[i] = (i < 35) ? be1[i] : 0.f;
    sb2[i]  = (i < 35) ? b2[i]  : 0.f;
    sg2[i]  = (i < 35) ? g2[i]  : 0.f;
    sbe2[i] = (i < 35) ? be2[i] : 0.f;
    if (i < 32) sb3[i] = b3[i];
  }
  __syncthreads();

  const int wave  = tid >> 5;
  const int lane  = tid & 31;
  const int tile  = blockIdx.x * 4 + wave;
  const int ebase = tile * 16;
  bf16*  X = sX[wave];
  float* H = sH[wave];

  // Stage input tile: cols 0..31 = m[dst] (gather), 32..34 = edge_feat, rest 0
  {
    const int row = lane & 15;
    const int e   = ebase + row;
    if ((lane >> 4) == 0) {
      const int dst = eidx[2 * e + 1];
      const float* mr = mbuf + (size_t)dst * D_MSG;
      for (int c = 0; c < 32; c += 2) {
        const v2bf pr = {(bf16)mr[c], (bf16)mr[c + 1]};
        *(v2bf*)(X + row * 64 + c) = pr;
      }
    } else {
      for (int c = 32; c < 64; c += 2) {
        const float f0 = (c < 35)     ? efeat[(size_t)e * 3 + (c - 32)]     : 0.f;
        const float f1 = (c + 1 < 35) ? efeat[(size_t)e * 3 + (c + 1 - 32)] : 0.f;
        const v2bf pr = {(bf16)f0, (bf16)f1};
        *(v2bf*)(X + row * 64 + c) = pr;
      }
    }
  }
  lds_fence();

  edge_layer(X, H, sW1, sb1, sg1, sbe1, lane);
  edge_layer(X, H, sW2, sb2, sg2, sbe2, lane);

  // Layer 3: 16x32 output, scatter-add to agg[src] and agg[dst]
  const v16bf a0 = frag_A(X, 64, 0);
  const v16bf a1 = frag_A(X, 64, 32);
#pragma unroll
  for (int nt = 0; nt < 2; ++nt) {
    v8f acc = zero8();
    acc = wmma_bf16(a0, frag_Bs(sW3, 0 * 2 + nt), acc);
    acc = wmma_bf16(a1, frag_Bs(sW3, 1 * 2 + nt), acc);
    const int col = nt * 16 + (lane & 15);
    const int rb  = (lane >> 4) << 3;
    const float bc = sb3[col];
#pragma unroll
    for (int r = 0; r < 8; ++r) {
      const int e  = ebase + rb + r;
      const float ph = acc[r] + bc;
      atomicAdd(&agg[(size_t)eidx[2 * e]     * D_MSG + col], ph);
      atomicAdd(&agg[(size_t)eidx[2 * e + 1] * D_MSG + col], ph);
    }
  }
}

// ---------------------------------------------------------------------------
// Build bf16 node features: [v,th,dP,dQ | m(32) | agg*Ainv(32) | pad->96]
// ---------------------------------------------------------------------------
__global__ void k_feats(const float* __restrict__ busf, const float* __restrict__ mbuf,
                        const float* __restrict__ agg, const float* __restrict__ ainv,
                        bf16* __restrict__ feats) {
  const int i = blockIdx.x * blockDim.x + threadIdx.x;   // N*96 threads
  const int n = i / 96, c = i % 96;
  float val;
  if      (c < 4)  val = busf[n * 4 + c];
  else if (c < 36) val = mbuf[(size_t)n * 32 + (c - 4)];
  else if (c < 68) val = agg[(size_t)n * 32 + (c - 36)] * ainv[n];
  else             val = 0.f;
  feats[i] = (bf16)val;
}

// ---------------------------------------------------------------------------
// Node MLP (din=dh=68 pad 96, dout 1 or 32).  64 threads = 2 waves, each
// owning a 16-node tile.  Layer-1 A tiles stream straight from global bf16
// features (global_load_b128 pairs); hidden layers via LDS + f32 LN.
// ---------------------------------------------------------------------------
__device__ __forceinline__ void node_layer(const bf16* Xin, bf16* Xout, float* H,
                                           const bf16* Wf, const float* bb,
                                           const float* gg, const float* bee, int lane) {
  const v16bf a0 = frag_A(Xin, 96, 0);
  const v16bf a1 = frag_A(Xin, 96, 32);
  const v16bf a2 = frag_A(Xin, 96, 64);
#pragma unroll
  for (int nt = 0; nt < 6; ++nt) {
    v8f acc = zero8();
    acc = wmma_bf16(a0, frag_Bs(Wf, 0 * 6 + nt), acc);
    acc = wmma_bf16(a1, frag_Bs(Wf, 1 * 6 + nt), acc);
    acc = wmma_bf16(a2, frag_Bs(Wf, 2 * 6 + nt), acc);
    const int col = nt * 16 + (lane & 15);
    const int rb  = (lane >> 4) << 3;
#pragma unroll
    for (int r = 0; r < 8; ++r) H[(rb + r) * 96 + col] = acc[r] + bb[col];
  }
  lds_fence();
  const int row = lane & 15;
  float s = 0.f, s2 = 0.f;
  for (int c = 0; c < 68; ++c) { const float x = H[row * 96 + c]; s += x; s2 += x * x; }
  const float mu  = s * (1.f / 68.f);
  const float var = s2 * (1.f / 68.f) - mu * mu;
  const float rs  = rsqrtf(var + 1e-5f);
  const int cb = (lane >> 4) * 48;
  for (int c = cb; c < cb + 48; c += 2) {
    float h0 = (c < 68)     ? ((H[row * 96 + c]     - mu) * rs * gg[c]     + bee[c])     : 0.f;
    float h1 = (c + 1 < 68) ? ((H[row * 96 + c + 1] - mu) * rs * gg[c + 1] + bee[c + 1]) : 0.f;
    h0 = (h0 > 0.f) ? h0 : 0.1f * h0;
    h1 = (h1 > 0.f) ? h1 : 0.1f * h1;
    const v2bf pr = {(bf16)h0, (bf16)h1};
    *(v2bf*)(Xout + row * 96 + c) = pr;
  }
  lds_fence();
}

__global__ __launch_bounds__(64) void k_node_mlp(
    const bf16* __restrict__ feats,
    const float* W1, const float* b1, const float* g1, const float* be1,
    const float* W2, const float* b2, const float* g2, const float* be2,
    const float* W3, const float* b3, float* __restrict__ out, int dout) {
  // Fragment-swizzled weights: [kc][nt][lane][16 elems]
  __shared__ __align__(32) bf16 sW1[3 * 6 * 32 * 16];   // 18 KB
  __shared__ __align__(32) bf16 sW2[3 * 6 * 32 * 16];   // 18 KB
  __shared__ __align__(32) bf16 sW3[3 * 2 * 32 * 16];   // 6 KB
  __shared__ float sb1[96], sg1[96], sbe1[96], sb2[96], sg2[96], sbe2[96], sb3[32];
  __shared__ __align__(32) bf16  sX[2][16 * 96];
  __shared__ __align__(32) float sH[2][16 * 96];

  const int tid = threadIdx.x;
  for (int i = tid; i < 3 * 6 * 512; i += 64) {
    const int e    = i & 15;
    const int ln   = (i >> 4) & 31;
    const int frag = i >> 9;                 // kc*6 + nt
    const int nt   = frag % 6;
    const int kc   = frag / 6;
    const int col  = nt * 16 + (ln & 15);
    const int k    = kc * 32 + ((ln & 16) ? 16 : 0) + e;
    const bool v   = (k < 68) && (col < 68);
    sW1[i] = v ? (bf16)W1[k * 68 + col] : (bf16)0.f;
    sW2[i] = v ? (bf16)W2[k * 68 + col] : (bf16)0.f;
  }
  for (int i = tid; i < 3 * 2 * 512; i += 64) {
    const int e    = i & 15;
    const int ln   = (i >> 4) & 31;
    const int frag = i >> 9;                 // kc*2 + nt
    const int col  = (frag & 1) * 16 + (ln & 15);
    const int k    = (frag >> 1) * 32 + ((ln & 16) ? 16 : 0) + e;
    sW3[i] = (k < 68 && col < dout) ? (bf16)W3[k * dout + col] : (bf16)0.f;
  }
  for (int i = tid; i < 96; i += 64) {
    sb1[i]  = (i < 68) ? b1[i]  : 0.f;
    sg1[i]  = (i < 68) ? g1[i]  : 0.f;
    sbe1[i] = (i < 68) ? be1[i] : 0.f;
    sb2[i]  = (i < 68) ? b2[i]  : 0.f;
    sg2[i]  = (i < 68) ? g2[i]  : 0.f;
    sbe2[i] = (i < 68) ? be2[i] : 0.f;
    if (i < 32) sb3[i] = (i < dout) ? b3[i] : 0.f;
  }
  __syncthreads();

  const int wave  = tid >> 5;
  const int lane  = tid & 31;
  const int tile  = blockIdx.x * 2 + wave;
  const int nbase = tile * 16;
  bf16*  X = sX[wave];
  float* H = sH[wave];

  node_layer(feats + (size_t)nbase * 96, X, H, sW1, sb1, sg1, sbe1, lane);
  node_layer(X, X, H, sW2, sb2, sg2, sbe2, lane);

  const v16bf a0 = frag_A(X, 96, 0);
  const v16bf a1 = frag_A(X, 96, 32);
  const v16bf a2 = frag_A(X, 96, 64);
  const int nt3 = (dout > 16) ? 2 : 1;
  for (int nt = 0; nt < nt3; ++nt) {
    v8f acc = zero8();
    acc = wmma_bf16(a0, frag_Bs(sW3, 0 * 2 + nt), acc);
    acc = wmma_bf16(a1, frag_Bs(sW3, 1 * 2 + nt), acc);
    acc = wmma_bf16(a2, frag_Bs(sW3, 2 * 2 + nt), acc);
    const int col = nt * 16 + (lane & 15);
    const int rb  = (lane >> 4) << 3;
#pragma unroll
    for (int r = 0; r < 8; ++r) {
      const int node = nbase + rb + r;
      const float val = acc[r] + sb3[col];
      if (dout == 1) { if (col == 0) out[node] = val; }
      else out[(size_t)node * 32 + col] = val;
    }
  }
}

// ---------------------------------------------------------------------------
// State update: mask dth/dv, tanh+normalize dm, wrap theta, clip v, m += dm
// ---------------------------------------------------------------------------
__global__ void k_update(const int* __restrict__ btype,
                         const float* __restrict__ outT, const float* __restrict__ outV,
                         const float* __restrict__ outM,
                         float* __restrict__ vst, float* __restrict__ thst,
                         float* __restrict__ mbuf) {
  const int n = blockIdx.x * blockDim.x + threadIdx.x;
  if (n >= N_NODES) return;
  const int bt = btype[n];
  const bool slack = (bt == 1);
  const bool sp    = (bt == 1) || (bt == 2);
  const float dth = slack ? 0.f : outT[n];
  const float dv  = sp    ? 0.f : outV[n];
  float dm[D_MSG];
  float mu = 0.f;
#pragma unroll
  for (int d = 0; d < D_MSG; ++d) { dm[d] = tanhf(outM[(size_t)n * D_MSG + d]); mu += dm[d]; }
  mu *= (1.f / D_MSG);
  float var = 0.f;
#pragma unroll
  for (int d = 0; d < D_MSG; ++d) { const float x = dm[d] - mu; var += x * x; }
  var *= (1.f / D_MSG);
  const float rs = rsqrtf(var + 1e-5f);
#pragma unroll
  for (int d = 0; d < D_MSG; ++d) mbuf[(size_t)n * D_MSG + d] += (dm[d] - mu) * rs;
  const float PI = 3.14159265358979323846f;
  float t = thst[n] + dth + PI;
  t = fmodf(t, 2.f * PI);
  if (t < 0.f) t += 2.f * PI;
  thst[n] = t - PI;
  const float v = vst[n] + dv;
  vst[n] = fminf(fmaxf(v, 0.4f), 1.2f);
}

__global__ void k_out(const float* __restrict__ vst, const float* __restrict__ thst,
                      const float* __restrict__ loss, float* __restrict__ out) {
  const int n = blockIdx.x * blockDim.x + threadIdx.x;
  if (n < N_NODES) { out[2 * n] = vst[n]; out[2 * n + 1] = thst[n]; }
  if (n == 0) out[2 * N_NODES] = *loss;
}

// ---------------------------------------------------------------------------
// Host orchestration (graph-capture safe: kernels + hipMemsetAsync only).
// Param leaves d_in[8..47] in jax pytree (sorted-key) order:
//   groups: edge, m, theta, v ; leaves: W1,W2,W3,b1,b2,b3,be1,be2,g1,g2
// ---------------------------------------------------------------------------
extern "C" void kernel_launch(void* const* d_in, const int* in_sizes, int n_in,
                              void* d_out, int out_size, void* d_ws, size_t ws_size,
                              hipStream_t stream) {
  (void)in_sizes; (void)n_in; (void)out_size; (void)ws_size;
  const int*   btype = (const int*)d_in[0];
  const float* Yr    = (const float*)d_in[1];
  const float* Yi    = (const float*)d_in[2];
  const float* Pset  = (const float*)d_in[3];
  const float* Qset  = (const float*)d_in[4];
  const float* V0    = (const float*)d_in[5];
  const int*   eidx  = (const int*)d_in[6];
  const float* efeat = (const float*)d_in[7];
  auto L = [&](int g, int i) { return (const float*)d_in[8 + g * 10 + i]; };

  // Workspace carve-up (all f32 except bf16 feats): ~2.5 MB total
  float* f = (float*)d_ws;
  float* vst  = f; f += N_NODES;
  float* thst = f; f += N_NODES;
  float* Vrb  = f; f += N_NODES;
  float* Vib  = f; f += N_NODES;
  float* busf = f; f += 4 * N_NODES;
  float* mbuf = f; f += D_MSG * N_NODES;
  float* agg  = f; f += D_MSG * N_NODES;
  float* outT = f; f += N_NODES;
  float* outV = f; f += N_NODES;
  float* outM = f; f += D_MSG * N_NODES;
  float* ainv = f; f += N_NODES;
  float* loss = f; f += 8;                 // padded so feats stays 32B-aligned
  bf16*  feats = (bf16*)f;                 // N*96 bf16

  hipMemsetAsync(mbuf, 0, (size_t)N_NODES * D_MSG * sizeof(float), stream);
  hipMemsetAsync(loss, 0, sizeof(float), stream);
  hipMemsetAsync(ainv, 0, (size_t)N_NODES * sizeof(float), stream);
  k_init<<<dim3(16), dim3(256), 0, stream>>>(V0, vst, thst);
  k_deg <<<dim3(N_EDGES / 256), dim3(256), 0, stream>>>(eidx, ainv);
  k_ainv<<<dim3(16), dim3(256), 0, stream>>>(ainv);

  for (int k = 0; k < K_STEPS; ++k) {
    const float w  = powf(0.96f, (float)(K_STEPS - 1 - k));
    const float wN = w / (float)N_NODES;

    k_polar<<<dim3(16), dim3(256), 0, stream>>>(vst, thst, Vrb, Vib);
    k_matvec<<<dim3(N_NODES), dim3(256), 0, stream>>>(
        Yr, Yi, Vrb, Vib, vst, thst, Pset, Qset, btype, busf, loss, wN);

    hipMemsetAsync(agg, 0, (size_t)N_NODES * D_MSG * sizeof(float), stream);
    k_edge_mlp<<<dim3(N_EDGES / 64), dim3(128), 0, stream>>>(
        mbuf, eidx, efeat,
        L(0,0) + k*1225, L(0,3) + k*35, L(0,8) + k*35, L(0,6) + k*35,
        L(0,1) + k*1225, L(0,4) + k*35, L(0,9) + k*35, L(0,7) + k*35,
        L(0,2) + k*1120, L(0,5) + k*32, agg);

    k_feats<<<dim3((N_NODES * 96) / 256), dim3(256), 0, stream>>>(
        busf, mbuf, agg, ainv, feats);

    // theta MLP (group 2, dout=1)
    k_node_mlp<<<dim3(N_NODES / 32), dim3(64), 0, stream>>>(
        feats,
        L(2,0) + k*4624, L(2,3) + k*68, L(2,8) + k*68, L(2,6) + k*68,
        L(2,1) + k*4624, L(2,4) + k*68, L(2,9) + k*68, L(2,7) + k*68,
        L(2,2) + k*68, L(2,5) + k, outT, 1);
    // v MLP (group 3, dout=1)
    k_node_mlp<<<dim3(N_NODES / 32), dim3(64), 0, stream>>>(
        feats,
        L(3,0) + k*4624, L(3,3) + k*68, L(3,8) + k*68, L(3,6) + k*68,
        L(3,1) + k*4624, L(3,4) + k*68, L(3,9) + k*68, L(3,7) + k*68,
        L(3,2) + k*68, L(3,5) + k, outV, 1);
    // m MLP (group 1, dout=32)
    k_node_mlp<<<dim3(N_NODES / 32), dim3(64), 0, stream>>>(
        feats,
        L(1,0) + k*4624, L(1,3) + k*68, L(1,8) + k*68, L(1,6) + k*68,
        L(1,1) + k*4624, L(1,4) + k*68, L(1,9) + k*68, L(1,7) + k*68,
        L(1,2) + k*2176, L(1,5) + k*32, outM, 32);

    k_update<<<dim3(N_NODES / 128), dim3(128), 0, stream>>>(
        btype, outT, outV, outM, vst, thst, mbuf);
  }

  k_out<<<dim3(16), dim3(256), 0, stream>>>(vst, thst, loss, (float*)d_out);
}
